// BitLinear_68118181314690
// MI455X (gfx1250) — 8-bit matrix operand order repaired and checked by reading, compiled, not run on hardware
//
#include <hip/hip_runtime.h>
#include <stdint.h>


typedef int v8i __attribute__((ext_vector_type(8)));

#define DEVINL __device__ __forceinline__


__global__ void k_init(uint32_t* s) {
    if (threadIdx.x == 0) s[0] = 0u;
}

__global__ void k_wabs(const float* __restrict__ w, float* __restrict__ part, int n) {
    __shared__ float red[256];
    float s = 0.f;
    for (int i = blockIdx.x * blockDim.x + threadIdx.x; i < n; i += gridDim.x * blockDim.x)
        s += fabsf(w[i]);
    red[threadIdx.x] = s;
    __syncthreads();
    for (int o = 128; o > 0; o >>= 1) {
        if (threadIdx.x < o) red[threadIdx.x] += red[threadIdx.x + o];
        __syncthreads();
    }
    if (threadIdx.x == 0) part[blockIdx.x] = red[0];
}

__global__ void k_rowstats(const float* __restrict__ x, const float* __restrict__ g,
                           float* __restrict__ rms, uint32_t* __restrict__ amax_bits, int K) {
    __shared__ float rs[256], rm[256];
    const float* row = x + (size_t)blockIdx.x * K;
    float ss = 0.f, mx = 0.f;
    for (int k = threadIdx.x * 4; k < K; k += blockDim.x * 4) {
        float4 v = *(const float4*)(row + k);
        float4 w = *(const float4*)(g + k);
        ss += v.x * v.x + v.y * v.y + v.z * v.z + v.w * v.w;
        mx = fmaxf(mx, fmaxf(fmaxf(fabsf(v.x * w.x), fabsf(v.y * w.y)),
                             fmaxf(fabsf(v.z * w.z), fabsf(v.w * w.w))));
    }
    rs[threadIdx.x] = ss;
    rm[threadIdx.x] = mx;
    __syncthreads();
    for (int o = 128; o > 0; o >>= 1) {
        if (threadIdx.x < o) {
            rs[threadIdx.x] += rs[threadIdx.x + o];
            rm[threadIdx.x] = fmaxf(rm[threadIdx.x], rm[threadIdx.x + o]);
        }
        __syncthreads();
    }
    if (threadIdx.x == 0) {
        float r = rsqrtf(rs[0] / (float)K + 1e-6f);
        rms[blockIdx.x] = r;
        float ra = fminf(r * rm[0], 1e4f);
        ra = (ra == ra) ? ra : 0.f;
        atomicMax(amax_bits, __float_as_uint(ra));
    }
}

__global__ void k_finalize(const float* __restrict__ part, int np, float* __restrict__ scal,
                           const uint32_t* __restrict__ amax_bits, float inv_nk) {
    __shared__ float red[256];
    float s = 0.f;
    for (int i = threadIdx.x; i < np; i += blockDim.x) s += part[i];
    red[threadIdx.x] = s;
    __syncthreads();
    for (int o = 128; o > 0; o >>= 1) {
        if (threadIdx.x < o) red[threadIdx.x] += red[threadIdx.x + o];
        __syncthreads();
    }
    if (threadIdx.x == 0) {
        float wsc = fmaxf(red[0] * inv_nk, 1e-4f);
        float asc = fmaxf(__uint_as_float(amax_bits[0]), 1e-5f);
        scal[4] = asc;
        scal[5] = wsc;
        scal[6] = 1.f / wsc;
        scal[7] = 127.f / asc;
        scal[8] = (asc / 127.f) * wsc;
    }
}

DEVINL int quant1(float xv, float gv, float r, float q) {
    float v = xv * r * gv;
    v = (v == v) ? v : 0.f;
    v = fminf(fmaxf(v, -1e4f), 1e4f);
    v = fminf(fmaxf(v * q, -128.f), 127.f);
    return (int)rintf(v);
}

__global__ void k_qx(const float* __restrict__ x, const float* __restrict__ g,
                     const float* __restrict__ rms, const float* __restrict__ scal,
                     uint8_t* __restrict__ xq, int K) {
    const int row = blockIdx.x;
    const float r = rms[row];
    const float q = scal[7];
    const float* xrow = x + (size_t)row * K;
    const int KT = K >> 6;
    const int mTile = row >> 4;
    const int rl = row & 15;
    uint32_t* base = (uint32_t*)xq;
    for (int k = threadIdx.x * 4; k < K; k += blockDim.x * 4) {
        float4 v = *(const float4*)(xrow + k);
        float4 w = *(const float4*)(g + k);
        int q0 = quant1(v.x, w.x, r, q), q1 = quant1(v.y, w.y, r, q);
        int q2 = quant1(v.z, w.z, r, q), q3 = quant1(v.w, w.w, r, q);
        uint32_t pack = (uint32_t)(q0 & 255) | ((uint32_t)(q1 & 255) << 8) |
                        ((uint32_t)(q2 & 255) << 16) | ((uint32_t)(q3 & 255) << 24);
        const int kt = k >> 6;
        const int d = (k >> 2) & 15;
        const int hi = (d >> 2) & 1;
        const int j = ((d >> 3) << 2) | (d & 3);
        const int lane = rl + (hi << 4);
        base[(size_t)(mTile * KT + kt) * 256 + lane * 8 + j] = pack;
    }
}

__global__ void k_qw(const float* __restrict__ w, const float* __restrict__ scal,
                     uint8_t* __restrict__ wq, int n) {
    float invw = scal[6];
    int i = (blockIdx.x * blockDim.x + threadIdx.x) * 4;
    if (i >= n) return;
    float4 v = *(const float4*)(w + i);
    int q0 = (int)rintf(fminf(fmaxf(v.x * invw, -1.f), 1.f));
    int q1 = (int)rintf(fminf(fmaxf(v.y * invw, -1.f), 1.f));
    int q2 = (int)rintf(fminf(fmaxf(v.z * invw, -1.f), 1.f));
    int q3 = (int)rintf(fminf(fmaxf(v.w * invw, -1.f), 1.f));
    ((uint32_t*)wq)[i >> 2] = (uint32_t)(q0 & 255) | ((uint32_t)(q1 & 255) << 8) |
                              ((uint32_t)(q2 & 255) << 16) | ((uint32_t)(q3 & 255) << 24);
}

#define WM_TILES 4
#define WN_TILES 2

template <int KC>
__global__ __launch_bounds__(256) void k_gemm(const uint8_t* __restrict__ xq,
                                              const uint8_t* __restrict__ wq,
                                              const float* __restrict__ scal,
                                              float* __restrict__ out,
                                              int M, int N, int Krt) {
    const int Kc = KC ? KC : Krt;
    const int KT = Kc >> 6;

    const int lane  = threadIdx.x & 31;
    const int wave  = threadIdx.x >> 5;
    const int waveM = wave >> 2;
    const int waveN = wave & 3;
    const int l15   = lane & 15;
    const int hi    = lane >> 4;

    const int m0 = blockIdx.y * 128 + waveM * 64;
    const int n0 = blockIdx.x * 128 + waveN * 32;

    v8i acc[WM_TILES][WN_TILES];
    const v8i zero = {0, 0, 0, 0, 0, 0, 0, 0};
#pragma unroll
    for (int i = 0; i < WM_TILES; ++i)
#pragma unroll
        for (int j = 0; j < WN_TILES; ++j) acc[i][j] = zero;

    const uint8_t* aBase = xq + ((size_t)(m0 >> 4) * KT) * 1024 + lane * 32;
    const uint8_t* bBase = wq + (size_t)(n0 + l15) * Kc + hi * 16;

    auto kstep = [&](int kt) {
        v8i b[WN_TILES];
#pragma unroll
        for (int j = 0; j < WN_TILES; ++j) {
            const uint8_t* bp = bBase + j * 16 * Kc + kt * 64;
            int4 t0 = *(const int4*)(bp);
            int4 t1 = *(const int4*)(bp + 32);
            b[j][0] = t0.x; b[j][1] = t0.y; b[j][2] = t0.z; b[j][3] = t0.w;
            b[j][4] = t1.x; b[j][5] = t1.y; b[j][6] = t1.z; b[j][7] = t1.w;
        }
#pragma unroll
        for (int i = 0; i < WM_TILES; ++i) {
            const uint8_t* ap = aBase + (i * KT + kt) * 1024;
            int4 t0 = *(const int4*)(ap);
            int4 t1 = *(const int4*)(ap + 16);
            v8i a;
            a[0] = t0.x; a[1] = t0.y; a[2] = t0.z; a[3] = t0.w;
            a[4] = t1.x; a[5] = t1.y; a[6] = t1.z; a[7] = t1.w;
#pragma unroll
            for (int j = 0; j < WN_TILES; ++j)
                acc[i][j] = __builtin_amdgcn_wmma_i32_16x16x64_iu8(
                     true, a,  true, b[j], acc[i][j],
                     false,  false);
        }
    };

    if constexpr (KC != 0) {
#pragma unroll
        for (int kt = 0; kt < KC / 64; ++kt) kstep(kt);
    } else {
        for (int kt = 0; kt < KT; ++kt) kstep(kt);
    }

    const float osc = scal[8];
#pragma unroll
    for (int i = 0; i < WM_TILES; ++i)
#pragma unroll
        for (int j = 0; j < WN_TILES; ++j)
#pragma unroll
            for (int e = 0; e < 8; ++e) {
                int m = m0 + i * 16 + hi * 8 + e;
                int n = n0 + j * 16 + l15;
                out[(size_t)m * N + n] = (float)acc[i][j][e] * osc;
            }
}

extern "C" void kernel_launch(void* const* d_in, const int* in_sizes, int n_in,
                              void* d_out, int out_size, void* d_ws, size_t ws_size,
                              hipStream_t stream) {
    (void)n_in; (void)out_size; (void)ws_size;
    const float* x = (const float*)d_in[0];
    const float* w = (const float*)d_in[1];
    const float* g = (const float*)d_in[2];
    float* out = (float*)d_out;

    const int K = in_sizes[2];
    const int N = in_sizes[1] / K;
    const int M = (int)((long long)in_sizes[0] / K);

    char* ws = (char*)d_ws;
    uint32_t* amax = (uint32_t*)ws;
    float* scal = (float*)ws;
    float* part = (float*)(ws + 256);
    float* rms = (float*)(ws + 4096);
    size_t off_wq = 4096 + (((size_t)M * 4 + 255) & ~(size_t)255);
    uint8_t* wq = (uint8_t*)(ws + off_wq);
    size_t off_xq = off_wq + (((size_t)N * K + 255) & ~(size_t)255);
    uint8_t* xq = (uint8_t*)(ws + off_xq);

    k_init<<<1, 64, 0, stream>>>(amax);
    k_wabs<<<512, 256, 0, stream>>>(w, part, N * K);
    k_rowstats<<<M, 256, 0, stream>>>(x, g, rms, amax, K);
    k_finalize<<<1, 256, 0, stream>>>(part, 512, scal, amax, 1.f / (float)(N * K));
    k_qx<<<M, 256, 0, stream>>>(x, g, rms, scal, xq, K);
    k_qw<<<(N * K) / 1024, 256, 0, stream>>>(w, scal, wq, N * K);

    dim3 grid(N / 128, M / 128);
    if (K == 1024)
        k_gemm<1024><<<grid, 256, 0, stream>>>(xq, wq, scal, out, M, N, K);
    else
        k_gemm<0><<<grid, 256, 0, stream>>>(xq, wq, scal, out, M, N, K);
}
